// MHA_9517647528204
// MI455X (gfx1250) — compile-verified, matrix operand order checked by reading, not run on hardware
//
#include <hip/hip_runtime.h>
#include <hip/hip_bf16.h>
#include <math.h>

// MI455X / gfx1250, wave32, WMMA f32_16x16x32_f16 + TDM (tensor_load_to_lds).
// B=2 S=2048 DIM=512 NH=8 QK=VD=128.

typedef _Float16 f16;
typedef __attribute__((ext_vector_type(16))) _Float16 v16h;
typedef __attribute__((ext_vector_type(8)))  float    v8f;
typedef unsigned int       u32;
typedef unsigned long long u64;
typedef u32 __attribute__((ext_vector_type(4))) v4u;
typedef int __attribute__((ext_vector_type(4))) v4i;
typedef int __attribute__((ext_vector_type(8))) v8i;

#define B_    2
#define S_    2048
#define DIM_  512
#define NH_   8
#define QK_   128
#define VD_   128
#define NEGBIG (-1.0e9f)

__device__ __forceinline__ v8f wmma16(v16h a, v16h b, v8f c) {
  return __builtin_amdgcn_wmma_f32_16x16x32_f16(false, a, false, b, (short)0, c,
                                                false, false);
}

__device__ __forceinline__ v8f zero8() {
  v8f z = {0.f, 0.f, 0.f, 0.f, 0.f, 0.f, 0.f, 0.f};
  return z;
}

// ---------------------------------------------------------------------------
// TDM: issue a 2D f16 tile DMA  global -> LDS  (Tensor DMA Descriptor per
// CDNA5 ISA ch.8: group0 = {count, lds_addr, global_addr, type=2},
// group1 = {data_size=2B, tensor_dim0/1, tile_dim0/1, tensor_dim0_stride}).
// Tile is written to LDS row-major, tile1 rows of tile0 elements.
// All dims/strides in f16 elements.  Tracked by TENSORcnt.
// ---------------------------------------------------------------------------
__device__ __forceinline__ void tdm_load_2d_f16(u32 lds_addr, const f16* gptr,
                                                u32 tile0, u32 tile1,
                                                u32 td0, u32 td1, u32 stride0) {
  const u64 ga = (u64)gptr;
  v4u g0;
  g0[0] = 1u;                                  // count=1, user descriptor
  g0[1] = lds_addr;                            // LDS byte address
  g0[2] = (u32)ga;                             // global_addr[31:0]
  g0[3] = (u32)(ga >> 32) | (2u << 30);        // global_addr[56:32] | type=2
  v8i g1;
  g1[0] = (int)(1u << 16);                     // data_size code 1 = 2 bytes
  g1[1] = (int)((td0 & 0xFFFFu) << 16);        // tensor_dim0[15:0]
  g1[2] = (int)(((td0 >> 16) & 0xFFFFu) | ((td1 & 0xFFFFu) << 16));
  g1[3] = (int)(((td1 >> 16) & 0xFFFFu) | (tile0 << 16));   // tile_dim0
  g1[4] = (int)tile1;                          // tile_dim1 (tile_dim2=0)
  g1[5] = (int)stride0;                        // tensor_dim0_stride[31:0]
  g1[6] = 0;
  g1[7] = 0;
  v4i gz = {0, 0, 0, 0};
#if defined(__clang_major__) && (__clang_major__ >= 23)
  v8i gz8 = {0, 0, 0, 0, 0, 0, 0, 0};
  __builtin_amdgcn_tensor_load_to_lds(g0, g1, gz, gz, gz8, 0);
#else
  __builtin_amdgcn_tensor_load_to_lds(g0, g1, gz, gz, 0);
#endif
}

__device__ __forceinline__ u32 lds_addr_of(const void* p) {
  return (u32)(u64)p;                          // low 32 bits = LDS offset
}

// 16x32 f16 fragment loader (A-layout; also serves B from [n][k] storage).
// element i -> K = (i&7) + (i>=8?16:0) + (lane>=16?8:0); pairs are K-adjacent
// so the compiler merges these into ds_load_b128s.
__device__ __forceinline__ v16h load_frag16(const f16* base, int ld) {
  const int lane = threadIdx.x & 31;
  const int row  = lane & 15;
  const int kh   = (lane >> 4) << 3;
  v16h r;
#pragma unroll
  for (int p = 0; p < 8; ++p) {
    const int k = ((p & 3) << 1) + ((p >> 2) << 4) + kh;
    r[2 * p]     = base[row * ld + k];
    r[2 * p + 1] = base[row * ld + k + 1];
  }
  return r;
}

// ---------------------------------------------------------------------------
// Kernel 1: fused Q/K/V projection.  C(4096x3072) = x(4096x512) @ [Wq|Wk|Wv]
// (staging converts f32->f16, so this one keeps VALU staging).
// Q,K stored [b][h][s][128]; V stored transposed [b][h][vd][S].
// ---------------------------------------------------------------------------
__global__ __launch_bounds__(256)
void qkv_proj_kernel(const float* __restrict__ x,
                     const float* __restrict__ wq_w, const float* __restrict__ wq_b,
                     const float* __restrict__ wk_w, const float* __restrict__ wk_b,
                     const float* __restrict__ wv_w, const float* __restrict__ wv_b,
                     f16* __restrict__ qbuf, f16* __restrict__ kbuf,
                     f16* __restrict__ vtbuf)
{
  __shared__ __align__(16) f16 Alds[128][32];
  __shared__ __align__(16) f16 Blds[128][32];

  const int t    = threadIdx.x;
  const int lane = t & 31;
  const int w    = t >> 5;
  const int r0   = blockIdx.x * 128;
  const int mat  = blockIdx.y >> 3;      // 0=q 1=k 2=v
  const int h    = blockIdx.y & 7;
  const int c0   = h * 128;

  const float* W    = (mat == 0) ? wq_w : (mat == 1) ? wk_w : wv_w;
  const float* bias = (mat == 0) ? wq_b : (mat == 1) ? wk_b : wv_b;

  const int wrow = (w & 3) * 32;
  const int wcol = (w >> 2) * 64;

  v8f acc[2][4];
#pragma unroll
  for (int i = 0; i < 2; ++i)
#pragma unroll
    for (int j = 0; j < 4; ++j) acc[i][j] = zero8();

  for (int kk = 0; kk < DIM_; kk += 32) {
    __syncthreads();
#pragma unroll
    for (int rep = 0; rep < 4; ++rep) {
      int idx = t + rep * 256;
      int row = idx >> 3;
      int c4  = (idx & 7) << 2;
      const float4 v = *(const float4*)(x + (size_t)(r0 + row) * DIM_ + kk + c4);
      Alds[row][c4 + 0] = (f16)v.x;
      Alds[row][c4 + 1] = (f16)v.y;
      Alds[row][c4 + 2] = (f16)v.z;
      Alds[row][c4 + 3] = (f16)v.w;
    }
    {
      const int n4 = (t & 31) * 4;
      const int kb = t >> 5;
#pragma unroll
      for (int rep = 0; rep < 4; ++rep) {
        int krow = kb + rep * 8;
        const float4 v = *(const float4*)(W + (size_t)(kk + krow) * 1024 + c0 + n4);
        Blds[n4 + 0][krow] = (f16)v.x;
        Blds[n4 + 1][krow] = (f16)v.y;
        Blds[n4 + 2][krow] = (f16)v.z;
        Blds[n4 + 3][krow] = (f16)v.w;
      }
    }
    __syncthreads();

    v16h a0 = load_frag16(&Alds[wrow][0], 32);
    v16h a1 = load_frag16(&Alds[wrow + 16][0], 32);
#pragma unroll
    for (int ni = 0; ni < 4; ++ni) {
      v16h bf = load_frag16(&Blds[wcol + ni * 16][0], 32);
      acc[0][ni] = wmma16(a0, bf, acc[0][ni]);
      acc[1][ni] = wmma16(a1, bf, acc[1][ni]);
    }
  }

  const int hh8 = (lane >> 4) << 3;
  const int n15 = lane & 15;
#pragma unroll
  for (int mi = 0; mi < 2; ++mi) {
#pragma unroll
    for (int ni = 0; ni < 4; ++ni) {
#pragma unroll
      for (int r = 0; r < 8; ++r) {
        const int M    = wrow + mi * 16 + r + hh8;
        const int grow = r0 + M;
        const int b    = grow >> 11;
        const int s    = grow & 2047;
        const int d    = wcol + ni * 16 + n15;
        const float val = acc[mi][ni][r] + bias[c0 + d];
        const size_t bh = (size_t)(b * NH_ + h);
        if (mat == 2) {
          vtbuf[(bh * VD_ + d) * S_ + s] = (f16)val;
        } else {
          f16* dst = (mat == 0) ? qbuf : kbuf;
          dst[(bh * S_ + s) * QK_ + d] = (f16)val;
        }
      }
    }
  }
}

// ---------------------------------------------------------------------------
// Kernel 2: causal flash attention, TDM-double-buffered K/V tiles.
// grid = (16 query-blocks of 128, 16 (b,h)); block = 256 (8 waves x 16 rows).
// ---------------------------------------------------------------------------
__global__ __launch_bounds__(256)
void attn_kernel(const f16* __restrict__ qbuf, const f16* __restrict__ kbuf,
                 const f16* __restrict__ vtbuf, const float* __restrict__ pos,
                 const int* __restrict__ start_pos, f16* __restrict__ attbuf)
{
  __shared__ __align__(16) f16 Klds[2][32][128];   // [buf][key][qk]
  __shared__ __align__(16) f16 Vlds[2][128][32];   // [buf][vd][key] (V^T tile)
  __shared__ __align__(16) f16 Plds[8][16][32];    // per-wave P scratch

  const int t    = threadIdx.x;
  const int lane = t & 31;
  const int w    = t >> 5;
  const int qb   = blockIdx.x;
  const int bh   = blockIdx.y;
  const int b    = bh >> 3;
  const int h    = bh & 7;
  const int sp   = start_pos[0];

  const int q0w  = qb * 128 + w * 16;
  const int hh8  = (lane >> 4) << 3;
  const int n15  = lane & 15;

  const f16* kbase = kbuf  + (size_t)bh * S_ * QK_;   // [key][128]
  const f16* vbase = vtbuf + (size_t)bh * VD_ * S_;   // [vd][S]

  // Q fragments (16 x 128) pinned in registers
  const f16* qbase = qbuf + ((size_t)bh * S_ + q0w) * QK_;
  v16h qa[4];
#pragma unroll
  for (int kt = 0; kt < 4; ++kt) qa[kt] = load_frag16(qbase + kt * 32, QK_);

  v8f oacc[8];
#pragma unroll
  for (int i = 0; i < 8; ++i) oacc[i] = zero8();
  float mrun[8], lrun[8];
#pragma unroll
  for (int r = 0; r < 8; ++r) { mrun[r] = -3.0e38f; lrun[r] = 0.f; }

  const float* pe     = pos + (size_t)h * S_ * S_;
  const int    nsteps = qb * 4 + 4;
  const float  scale  = 0.08838834764831845f;   // 1/sqrt(128)

  // Prologue: wave 0 launches the TDM pair for step 0 into buffer 0.
  if (w == 0) {
    tdm_load_2d_f16(lds_addr_of(&Klds[0][0][0]), kbase, 128, 32, 128, 32, 128);
    tdm_load_2d_f16(lds_addr_of(&Vlds[0][0][0]), vbase, 32, 128, 32, 128, S_);
  }

  for (int step = 0; step < nsteps; ++step) {
    const int k0  = step * 32;
    const int cur = step & 1;
    if (w == 0) {
      if (step + 1 < nsteps) {     // prefetch next tiles into the other buffer
        const int k1 = k0 + 32, nb = cur ^ 1;
        tdm_load_2d_f16(lds_addr_of(&Klds[nb][0][0]),
                        kbase + (size_t)k1 * QK_, 128, 32, 128, 32, 128);
        tdm_load_2d_f16(lds_addr_of(&Vlds[nb][0][0]),
                        vbase + k1, 32, 128, 32, 128, S_);
        __builtin_amdgcn_s_wait_tensorcnt(2);   // current pair done, next in flight
      } else {
        __builtin_amdgcn_s_wait_tensorcnt(0);
      }
    }
    __syncthreads();   // publish TDM-written LDS to all waves

    if (k0 <= q0w + 15) {      // wave-uniform causal skip (EXEC stays all-1s)
      // ---- scores: S(16x32) = Q(16x128) @ K^T ----
      v8f s0 = zero8(), s1 = zero8();
#pragma unroll
      for (int kt = 0; kt < 4; ++kt) {
        v16h kb0 = load_frag16(&Klds[cur][0][0]  + kt * 32, QK_);
        v16h kb1 = load_frag16(&Klds[cur][16][0] + kt * 32, QK_);
        s0 = wmma16(qa[kt], kb0, s0);
        s1 = wmma16(qa[kt], kb1, s1);
      }

      // ---- scale + positional bias + causal mask, online softmax ----
      float a0v[8], a1v[8], mnew[8];
#pragma unroll
      for (int r = 0; r < 8; ++r) {
        const int m = r + hh8;
        const int q = q0w + m;
        const float* perow = pe + (size_t)(sp + q) * S_ + sp + k0;
        float a0 = s0[r] * scale + perow[n15]
                   + ((k0 + n15)      <= q ? 0.f : NEGBIG);
        float a1 = s1[r] * scale + perow[16 + n15]
                   + ((k0 + 16 + n15) <= q ? 0.f : NEGBIG);
        __builtin_prefetch((const void*)(perow + 32), 0, 0);
        a0v[r] = a0; a1v[r] = a1;
        float mx = fmaxf(a0, a1);
        mx = fmaxf(mx, __shfl_xor(mx, 1, 32));
        mx = fmaxf(mx, __shfl_xor(mx, 2, 32));
        mx = fmaxf(mx, __shfl_xor(mx, 4, 32));
        mx = fmaxf(mx, __shfl_xor(mx, 8, 32));
        mnew[r] = fmaxf(mrun[r], mx);
      }
#pragma unroll
      for (int r = 0; r < 8; ++r) {
        float e0 = __expf(a0v[r] - mnew[r]);
        float e1 = __expf(a1v[r] - mnew[r]);
        float sum = e0 + e1;
        sum += __shfl_xor(sum, 1, 32);
        sum += __shfl_xor(sum, 2, 32);
        sum += __shfl_xor(sum, 4, 32);
        sum += __shfl_xor(sum, 8, 32);
        const float corr = __expf(mrun[r] - mnew[r]);
        lrun[r] = lrun[r] * corr + sum;
        mrun[r] = mnew[r];
#pragma unroll
        for (int tt = 0; tt < 8; ++tt) oacc[tt][r] *= corr;
        const int m = r + hh8;
        Plds[w][m][n15]      = (f16)e0;
        Plds[w][m][16 + n15] = (f16)e1;
      }

      // ---- O(16x128) += P(16x32) @ V(32x128) ----
      v16h pf = load_frag16(&Plds[w][0][0], 32);
#pragma unroll
      for (int tt = 0; tt < 8; ++tt) {
        v16h vf = load_frag16(&Vlds[cur][tt * 16][0], 32);
        oacc[tt] = wmma16(pf, vf, oacc[tt]);
      }
    }
    __syncthreads();   // all reads of buf[cur] done before it is re-targeted
  }

  // epilogue: normalize and store [b][s][h*128+vd]
#pragma unroll
  for (int r = 0; r < 8; ++r) {
    const int m = r + hh8;
    const int q = q0w + m;
    const float inv = 1.0f / lrun[r];
    const size_t ob = (size_t)(b * S_ + q) * (NH_ * VD_) + h * VD_;
#pragma unroll
    for (int tt = 0; tt < 8; ++tt)
      attbuf[ob + tt * 16 + n15] = (f16)(oacc[tt][r] * inv);
  }
}

// ---------------------------------------------------------------------------
// Kernel 3: output projection.  out(4096x512) = att(4096x1024) @ wo + wo_b
// A-tile is a pure f16 tile copy -> staged by TDM.
// ---------------------------------------------------------------------------
__global__ __launch_bounds__(256)
void out_proj_kernel(const f16* __restrict__ attbuf,
                     const float* __restrict__ wo_w,
                     const float* __restrict__ wo_b,
                     float* __restrict__ out)
{
  __shared__ __align__(16) f16 Alds[128][32];
  __shared__ __align__(16) f16 Blds[128][32];

  const int t    = threadIdx.x;
  const int lane = t & 31;
  const int w    = t >> 5;
  const int r0   = blockIdx.x * 128;
  const int c0   = blockIdx.y * 128;
  const int wrow = (w & 3) * 32;
  const int wcol = (w >> 2) * 64;

  v8f acc[2][4];
#pragma unroll
  for (int i = 0; i < 2; ++i)
#pragma unroll
    for (int j = 0; j < 4; ++j) acc[i][j] = zero8();

  for (int kk = 0; kk < NH_ * VD_; kk += 32) {
    __syncthreads();
    if (w == 0) {   // TDM: att[r0..r0+127][kk..kk+31] -> Alds, 128 rows x 32 f16
      tdm_load_2d_f16(lds_addr_of(&Alds[0][0]),
                      attbuf + (size_t)r0 * 1024 + kk, 32, 128, 32, 128, 1024);
    }
    {   // stage B transposed: wo_w[kk+k][c0+n] -> Blds[n][k]  (f32 -> f16)
      const int n4 = (t & 31) * 4;
      const int kb = t >> 5;
#pragma unroll
      for (int rep = 0; rep < 4; ++rep) {
        int krow = kb + rep * 8;
        const float4 v = *(const float4*)(wo_w + (size_t)(kk + krow) * 512 + c0 + n4);
        Blds[n4 + 0][krow] = (f16)v.x;
        Blds[n4 + 1][krow] = (f16)v.y;
        Blds[n4 + 2][krow] = (f16)v.z;
        Blds[n4 + 3][krow] = (f16)v.w;
      }
    }
    if (w == 0) __builtin_amdgcn_s_wait_tensorcnt(0);
    __syncthreads();

    v16h a0 = load_frag16(&Alds[wrow][0], 32);
    v16h a1 = load_frag16(&Alds[wrow + 16][0], 32);
#pragma unroll
    for (int ni = 0; ni < 4; ++ni) {
      v16h bf = load_frag16(&Blds[wcol + ni * 16][0], 32);
      acc[0][ni] = wmma16(a0, bf, acc[0][ni]);
      acc[1][ni] = wmma16(a1, bf, acc[1][ni]);
    }
  }

  const int hh8 = (lane >> 4) << 3;
  const int n15 = lane & 15;
#pragma unroll
  for (int mi = 0; mi < 2; ++mi) {
#pragma unroll
    for (int ni = 0; ni < 4; ++ni) {
#pragma unroll
      for (int r = 0; r < 8; ++r) {
        const int M    = wrow + mi * 16 + r + hh8;
        const int grow = r0 + M;
        const int col  = c0 + wcol + ni * 16 + n15;
        out[(size_t)grow * 512 + col] = acc[mi][ni][r] + wo_b[col];
      }
    }
  }
}

// ---------------------------------------------------------------------------
extern "C" void kernel_launch(void* const* d_in, const int* in_sizes, int n_in,
                              void* d_out, int out_size, void* d_ws, size_t ws_size,
                              hipStream_t stream) {
  (void)in_sizes; (void)n_in; (void)out_size; (void)ws_size;
  const float* x    = (const float*)d_in[0];
  const int*   sp   = (const int*)d_in[1];
  // d_in[2] = mask: unused — causal mask computed analytically.
  const float* pos  = (const float*)d_in[3];
  const float* wq_w = (const float*)d_in[4];
  const float* wq_b = (const float*)d_in[5];
  const float* wk_w = (const float*)d_in[6];
  const float* wk_b = (const float*)d_in[7];
  const float* wv_w = (const float*)d_in[8];
  const float* wv_b = (const float*)d_in[9];
  const float* wo_w = (const float*)d_in[10];
  const float* wo_b = (const float*)d_in[11];

  f16* ws = (f16*)d_ws;
  const size_t per = (size_t)B_ * NH_ * S_ * 128;   // 4,194,304 halves
  f16* qbuf  = ws;                // [b][h][s][128]
  f16* kbuf  = ws + per;          // [b][h][s][128]
  f16* vtbuf = ws + 2 * per;      // [b][h][vd][S]
  f16* attb  = ws + 3 * per;      // [b][s][h*128+vd]

  qkv_proj_kernel<<<dim3(32, 24), 256, 0, stream>>>(
      x, wq_w, wq_b, wk_w, wk_b, wv_w, wv_b, qbuf, kbuf, vtbuf);
  attn_kernel<<<dim3(16, 16), 256, 0, stream>>>(
      qbuf, kbuf, vtbuf, pos, sp, attb);
  out_proj_kernel<<<dim3(32, 4), 256, 0, stream>>>(
      attb, wo_w, wo_b, (float*)d_out);
}
